// Head_18004502905076
// MI455X (gfx1250) — compile-verified, matrix operand order checked by reading, not run on hardware
//
#include <hip/hip_runtime.h>
#include <hip/hip_bf16.h>
#include <math.h>

#define BATCH 4
#define TLEN  4096
#define EMBED 512
#define HEAD  64

typedef __attribute__((ext_vector_type(16))) _Float16 v16h;
typedef __attribute__((ext_vector_type(8)))  _Float16 v8h;
typedef __attribute__((ext_vector_type(4)))  _Float16 v4h;
typedef __attribute__((ext_vector_type(8)))  float    v8f;
typedef __attribute__((ext_vector_type(4)))  float    v4f;

// ---- WMMA helper: D = A*B + C, f16 inputs, f32 accumulate -------------------
__device__ __forceinline__ v8f wmma_f16(v16h a, v16h b, v8f c) {
    return __builtin_amdgcn_wmma_f32_16x16x32_f16(
        /*neg_a=*/false, a, /*neg_b=*/false, b,
        /*c_mod=*/(short)0, c, /*reuse_a=*/false, /*reuse_b=*/false);
}

// ---- A/B fragment loaders (16x32 tile, K along contiguous axis) -------------
// ISA layout (16-bit A 16x32): lane L holds row M=L%16;
//   halves 0..7  = K in [ko,   ko+7]   where ko = (L/16)*8
//   halves 8..15 = K in [ko+16,ko+23]
__device__ __forceinline__ v16h load_frag_f16(const _Float16* __restrict__ p, int stride) {
    int lane = threadIdx.x & 31;
    int r  = lane & 15;
    int ko = (lane >> 4) << 3;
    union { v16h v; v8h h[2]; } u;
    u.h[0] = *reinterpret_cast<const v8h*>(p + (size_t)r * stride + ko);
    u.h[1] = *reinterpret_cast<const v8h*>(p + (size_t)r * stride + ko + 16);
    return u.v;
}

__device__ __forceinline__ v16h load_frag_f32(const float* __restrict__ p, int stride) {
    int lane = threadIdx.x & 31;
    int r  = lane & 15;
    int ko = (lane >> 4) << 3;
    const v4f* a0 = reinterpret_cast<const v4f*>(p + (size_t)r * stride + ko);
    const v4f* a1 = reinterpret_cast<const v4f*>(p + (size_t)r * stride + ko + 16);
    v4f x0 = a0[0], x1 = a0[1], x2 = a1[0], x3 = a1[1];
    union { v16h v; _Float16 h[16]; } u;
#pragma unroll
    for (int i = 0; i < 4; ++i) {
        u.h[i]      = (_Float16)x0[i];
        u.h[4 + i]  = (_Float16)x1[i];
        u.h[8 + i]  = (_Float16)x2[i];
        u.h[12 + i] = (_Float16)x3[i];
    }
    return u.v;
}

// ---- Kernel 0: one-time fp32 -> f16 conversion of the three weight matrices -
// grid = (EMBED*HEAD/(256*4), 3), block = 256; w16 layout: [3][HEAD][EMBED]
__global__ void wconv_kernel(const float* __restrict__ Wk,
                             const float* __restrict__ Wq,
                             const float* __restrict__ Wv,
                             _Float16* __restrict__ w16) {
    const int p = blockIdx.y;
    const float* src = (p == 0) ? Wk : (p == 1) ? Wq : Wv;
    const int i = (blockIdx.x * blockDim.x + threadIdx.x) * 4;
    v4f v = *reinterpret_cast<const v4f*>(src + i);
    v4h h;
#pragma unroll
    for (int e = 0; e < 4; ++e) h[e] = (_Float16)v[e];
    *reinterpret_cast<v4h*>(w16 + (size_t)p * HEAD * EMBED + i) = h;
}

// ---- Kernel 1: fused projections k/q/v from one x read ----------------------
// grid = BATCH*TLEN/16, block = 32. One wave: 16 rows of x -> 16x64 k, q, vT.
__global__ void proj_kernel(const float* __restrict__ x,
                            const _Float16* __restrict__ w16,
                            _Float16* __restrict__ k16,
                            _Float16* __restrict__ q16,
                            _Float16* __restrict__ vT16) {
    const int tile = blockIdx.x;
    const int b  = tile >> 8;          // TLEN/16 = 256 tiles per batch
    const int t0 = (tile & 255) << 4;
    const float* xb = x + ((size_t)b * TLEN + t0) * EMBED;

    v8f acc[3][4];
#pragma unroll
    for (int p = 0; p < 3; ++p)
#pragma unroll
        for (int h = 0; h < 4; ++h) acc[p][h] = v8f{0};

    for (int c0 = 0; c0 < EMBED; c0 += 32) {
        v16h a = load_frag_f32(xb + c0, EMBED);   // converted once, used 12x
#pragma unroll
        for (int p = 0; p < 3; ++p) {
            // batch the 4 B-fragment loads ahead of the 4 WMMAs so the
            // scheduler overlaps p+1's loads with p's matrix ops
            v16h bf[4];
#pragma unroll
            for (int h = 0; h < 4; ++h)
                bf[h] = load_frag_f16(w16 + ((size_t)p * HEAD + h * 16) * EMBED + c0, EMBED);
#pragma unroll
            for (int h = 0; h < 4; ++h)
                acc[p][h] = wmma_f16(a, bf[h], acc[p][h]);
        }
    }

    // C/D layout: lane L, vgpr j -> element (M = j + 8*(L/16), N = L%16)
    const int lane = threadIdx.x & 31;
    const int n  = lane & 15;
    const int mo = (lane >> 4) << 3;

    _Float16* kb = k16 + ((size_t)b * TLEN + t0) * HEAD;
    _Float16* qb = q16 + ((size_t)b * TLEN + t0) * HEAD;
    _Float16* vb = vT16 + (size_t)b * HEAD * TLEN;   // v stored transposed [H][T]
#pragma unroll
    for (int h = 0; h < 4; ++h)
#pragma unroll
        for (int j = 0; j < 8; ++j) {
            kb[(size_t)(mo + j) * HEAD + h * 16 + n] = (_Float16)acc[0][h][j];
            qb[(size_t)(mo + j) * HEAD + h * 16 + n] = (_Float16)acc[1][h][j];
            vb[(size_t)(h * 16 + n) * TLEN + t0 + mo + j] = (_Float16)acc[2][h][j];
        }
}

// ---- Kernel 2: causal flash attention, 64-wide KV blocks --------------------
// grid = BATCH*TLEN/16, block = 32 (one wave owns 16 query rows x 64 head dims)
__global__ void attn_kernel(const _Float16* __restrict__ q16,
                            const _Float16* __restrict__ k16,
                            const _Float16* __restrict__ vT16,
                            float* __restrict__ out) {
    __shared__ _Float16 ptile[16 * 64];   // P staging: D-layout -> A-layout

    const int tile = blockIdx.x;
    const int b  = tile >> 8;
    const int qt = tile & 255;
    const int t0 = qt << 4;

    const _Float16* qb = q16 + ((size_t)b * TLEN + t0) * HEAD;
    const _Float16* kb = k16 + (size_t)b * TLEN * HEAD;
    const _Float16* vb = vT16 + (size_t)b * HEAD * TLEN;

    const int lane = threadIdx.x & 31;
    const int n  = lane & 15;
    const int mo = (lane >> 4) << 3;

    // Q fragments for h-chunks [0,32) and [32,64), loaded once
    const v16h qf0 = load_frag_f16(qb, HEAD);
    const v16h qf1 = load_frag_f16(qb + 32, HEAD);

    v8f o[4] = {v8f{0}, v8f{0}, v8f{0}, v8f{0}};
    float m[8], l[8];
#pragma unroll
    for (int j = 0; j < 8; ++j) { m[j] = -INFINITY; l[j] = 0.0f; }

    const float scale = 0.04419417382415922f;   // 512^-0.5
    const int send = t0 + 16;                    // iterate s0 < send, step 64
    // (s0 is always <= t0, so every row has >= 1 unmasked column per block)

    for (int s0 = 0; s0 < send; s0 += 64) {
        // --- load all K fragments, then S = Q K^T (8 WMMAs) ---
        v16h kf[8];
#pragma unroll
        for (int hh = 0; hh < 4; ++hh) {
            const _Float16* kp = kb + (size_t)(s0 + 16 * hh) * HEAD;
            kf[2 * hh]     = load_frag_f16(kp, HEAD);       // B^T = K rows, h 0..31
            kf[2 * hh + 1] = load_frag_f16(kp + 32, HEAD);  // h 32..63
        }
        v8f s[4];
#pragma unroll
        for (int hh = 0; hh < 4; ++hh) {
            s[hh] = wmma_f16(qf0, kf[2 * hh], v8f{0});
            s[hh] = wmma_f16(qf1, kf[2 * hh + 1], s[hh]);
        }

        // --- hoist V fragment loads: independent of softmax, latency hidden ---
        v16h vf[8];
#pragma unroll
        for (int h = 0; h < 4; ++h) {
            vf[2 * h]     = load_frag_f16(vb + (size_t)(h * 16) * TLEN + s0,      TLEN);
            vf[2 * h + 1] = load_frag_f16(vb + (size_t)(h * 16) * TLEN + s0 + 32, TLEN);
        }

        // --- online softmax update, per row j (row M = mo + j) ---
#pragma unroll
        for (int j = 0; j < 8; ++j) {
            const int trow = t0 + mo + j;
            float best = -INFINITY;
#pragma unroll
            for (int hh = 0; hh < 4; ++hh) {
                const int scol = s0 + 16 * hh + n;
                float val = s[hh][j] * scale;
                if (scol > trow) val = -INFINITY;
                s[hh][j] = val;
                best = fmaxf(best, val);
            }
#pragma unroll
            for (int d = 1; d < 16; d <<= 1)
                best = fmaxf(best, __shfl_xor(best, d, 32));
            const float mn = fmaxf(m[j], best);
            const float alpha = __expf(m[j] - mn);     // first iter: exp(-inf)=0
            float rs = 0.0f;
#pragma unroll
            for (int hh = 0; hh < 4; ++hh) {
                float p = __expf(s[hh][j] - mn);       // masked: exp(-inf)=0
                s[hh][j] = p;
                rs += p;
            }
#pragma unroll
            for (int d = 1; d < 16; d <<= 1)
                rs += __shfl_xor(rs, d, 32);
            l[j] = l[j] * alpha + rs;
            m[j] = mn;
            o[0][j] *= alpha; o[1][j] *= alpha; o[2][j] *= alpha; o[3][j] *= alpha;
        }

        // --- stage P through LDS: D-layout -> row-major 16x64 f16 ---
#pragma unroll
        for (int j = 0; j < 8; ++j)
#pragma unroll
            for (int hh = 0; hh < 4; ++hh)
                ptile[(mo + j) * 64 + 16 * hh + n] = (_Float16)s[hh][j];
        asm volatile("s_wait_dscnt 0x0" ::: "memory");  // intra-wave LDS RAW

        v16h pf0 = load_frag_f16(&ptile[0],  64);       // P cols [0,32)
        v16h pf1 = load_frag_f16(&ptile[32], 64);       // P cols [32,64)

        // --- O += P * V (8 WMMAs; V fragments already resident) ---
#pragma unroll
        for (int h = 0; h < 4; ++h) {
            o[h] = wmma_f16(pf0, vf[2 * h], o[h]);
            o[h] = wmma_f16(pf1, vf[2 * h + 1], o[h]);
        }
    }

    // --- normalize and write out [B][T][H] fp32 ---
    float* ob = out + ((size_t)b * TLEN + t0) * HEAD;
#pragma unroll
    for (int h = 0; h < 4; ++h)
#pragma unroll
        for (int j = 0; j < 8; ++j)
            ob[(size_t)(mo + j) * HEAD + h * 16 + n] = o[h][j] / l[j];
}

extern "C" void kernel_launch(void* const* d_in, const int* in_sizes, int n_in,
                              void* d_out, int out_size, void* d_ws, size_t ws_size,
                              hipStream_t stream) {
    (void)in_sizes; (void)n_in; (void)out_size; (void)ws_size;
    const float* x  = (const float*)d_in[0];
    const float* Wk = (const float*)d_in[1];
    const float* Wq = (const float*)d_in[2];
    const float* Wv = (const float*)d_in[3];
    float* out = (float*)d_out;

    const size_t per = (size_t)BATCH * TLEN * HEAD;    // 1 Mi elements
    _Float16* k16  = (_Float16*)d_ws;
    _Float16* q16  = k16 + per;
    _Float16* vT16 = q16 + per;
    _Float16* w16  = vT16 + per;                       // 3*HEAD*EMBED halves

    // weights: 3 * 64*512 elements, 4 per thread
    wconv_kernel<<<dim3(HEAD * EMBED / (256 * 4), 3), 256, 0, stream>>>(Wk, Wq, Wv, w16);

    const int tiles = BATCH * TLEN / 16;               // 1024
    proj_kernel<<<dim3(tiles), 32, 0, stream>>>(x, w16, k16, q16, vT16);
    attn_kernel<<<dim3(tiles), 32, 0, stream>>>(q16, k16, vT16, out);
}
